// TimeframeSelector_72662256714498
// MI455X (gfx1250) — matrix operand order repaired and checked by reading, not compiled, not run on hardware
//
#include <hip/hip_runtime.h>
#include <hip/hip_bf16.h>
#include <math.h>

#define DEVFN __device__ __forceinline__

typedef __attribute__((ext_vector_type(16))) __bf16 bf16x16;
typedef __attribute__((ext_vector_type(8)))  __bf16 bf16x8;
typedef __attribute__((ext_vector_type(4)))  __bf16 bf16x4;
typedef __attribute__((ext_vector_type(8)))  float  f32x8;

constexpr int Bsz = 65536;
constexpr int D   = 256;
constexpr int T   = 5;
constexpr int R   = 6;
constexpr int S   = 3;
constexpr int TD  = 1280;   // T*D
constexpr int H1  = 512;    // 2*D
constexpr int H2  = 128;    // D/2

constexpr int MT    = 64;   // batch rows per workgroup
constexpr int NTHR  = 256;  // 8 wave32s

// LDS row strides in bf16 elements (rows stay 16B aligned)
constexpr int XS = TD + 8;  // 1288
constexpr int HS = H1 + 8;  // 520
constexpr int AS = D  + 8;  // 264
constexpr int TS = H2 + 8;  // 136

constexpr int OFF_X  = 0;
constexpr int OFF_H  = OFF_X  + MT * XS * 2;  // 164864
constexpr int OFF_A  = OFF_H  + MT * HS * 2;  // 231424
constexpr int OFF_T1 = OFF_A  + MT * AS * 2;  // 265216
constexpr int OFF_ST = OFF_T1 + MT * TS * 2;  // 282624
constexpr int OFF_PR = OFF_ST + 2 * MT * 4;   // 283136
constexpr int OFF_W  = OFF_PR + MT * 8 * 4;   // 285184
constexpr int OFF_Q  = OFF_W  + MT * 8 * 4;   // 287232
constexpr int OFF_TMP = OFF_Q + MT * 8 * 4;   // 289280
constexpr int SMEM_BYTES = OFF_TMP + MT * 8 * 4; // 291328  (< 320KB WGP LDS)

DEVFN f32x8 zero8() {
    f32x8 v;
#pragma unroll
    for (int i = 0; i < 8; ++i) v[i] = 0.f;
    return v;
}
DEVFN float gelu_erf(float x) { return 0.5f * x * (1.f + erff(x * 0.70710678118654752f)); }
DEVFN float sigmoidf(float x) { return 1.f / (1.f + expf(-x)); }

// A fragment (16x32 bf16): lane l holds row l%16; VGPR0..3 = K base..base+7
// (base = (l/16)*8), VGPR4..7 = K base+16..base+23.
DEVFN bf16x16 load_frag_a(const __bf16* p0, int stride, int lane) {
    const __bf16* p = p0 + (lane & 15) * stride + ((lane >> 4) << 3);
    union { bf16x16 v; bf16x8 h[2]; } u;
    u.h[0] = *(const bf16x8*)(p);
    u.h[1] = *(const bf16x8*)(p + 16);
    return u.v;
}
// B fragment (32x16 bf16) from W^T stored [N][K]: lane l holds column n0+l%16,
// K k0+base..base+7 then k0+base+16..base+23 (base = (l/16)*8), matching A.
DEVFN bf16x16 load_frag_b(const __bf16* wt, int kstride, int n0, int k0, int lane) {
    const __bf16* p = wt + (size_t)(n0 + (lane & 15)) * kstride + k0 + ((lane >> 4) << 3);
    union { bf16x16 v; bf16x8 h[2]; } u;
    u.h[0] = *(const bf16x8*)(p);
    u.h[1] = *(const bf16x8*)(p + 16);
    return u.v;
}
#define WMMA_BF16(a, b, c) \
    __builtin_amdgcn_wmma_f32_16x16x32_bf16(false, (a), false, (b), (short)0, (c), false, false)

struct KParams {
    const float *tf, *enc_b1, *ln_g, *ln_b, *enc_b2;
    const float *head_b1, *head_w2, *head_b2, *regime_emb;
    const float *q_b1, *q_w2, *q_b2;
    const float *ref_w1, *ref_b1, *ref_w2, *ref_b2, *gate_w, *gate_b, *step_emb;
    const float *regime_priors, *final_w, *final_b;
    const __bf16 *W1t, *W2t, *HW1t, *QW1t, *RW1t;
    float* out;
};

__global__ void __launch_bounds__(NTHR)
fused_tfsel(KParams p)
{
    extern __shared__ char smem[];
    __bf16* sX   = (__bf16*)(smem + OFF_X);
    __bf16* sH   = (__bf16*)(smem + OFF_H);
    __bf16* sA   = (__bf16*)(smem + OFF_A);   // aggregated, later combined
    __bf16* sT1  = (__bf16*)(smem + OFF_T1);
    float*  sSum = (float*)(smem + OFF_ST);
    float*  sSq  = sSum + MT;
    float*  sPr  = (float*)(smem + OFF_PR);
    float*  sW   = (float*)(smem + OFF_W);
    float*  sQ   = (float*)(smem + OFF_Q);
    float*  sTmp = (float*)(smem + OFF_TMP);

    const int tid  = threadIdx.x;
    const int lane = tid & 31;
    const int wv   = tid >> 5;
    const int col  = lane & 15;
    const int hsel = lane >> 4;
    const int b0   = blockIdx.x * MT;

    // ---- Phase 1: stage X tile (64 x 1280) into LDS as bf16 (single HBM pass) ----
    // Vectorized b128 loads / 8B LDS stores: this phase sets the 23.3 TB/s roofline.
    {
        constexpr int TD4 = TD / 4;
        for (int idx = tid; idx < MT * TD4; idx += NTHR) {
            int r = idx / TD4, c = (idx - r * TD4) * 4;
            int t = c >> 8, d = c & (D - 1);
            const float* src = p.tf + (size_t)t * Bsz * D + (size_t)(b0 + r) * D + d;
            float4 v = *(const float4*)src;
            bf16x4 o;
            o[0] = (__bf16)v.x; o[1] = (__bf16)v.y; o[2] = (__bf16)v.z; o[3] = (__bf16)v.w;
            *(bf16x4*)(sX + r * XS + c) = o;
        }
    }
    if (tid < MT) { sSum[tid] = 0.f; sSq[tid] = 0.f; }
    __syncthreads();

    // ---- Phase 2: h = gelu(LN(X @ W1 + b1)); M=64 N=512 K=1280 ----
    {
        f32x8 acc[4][4];
#pragma unroll
        for (int m = 0; m < 4; ++m)
#pragma unroll
            for (int n = 0; n < 4; ++n) acc[m][n] = zero8();

        for (int kt = 0; kt < TD / 32; ++kt) {
            const int k0 = kt * 32;
            bf16x16 a[4], b[4];
#pragma unroll
            for (int m = 0; m < 4; ++m)
                a[m] = load_frag_a(sX + (m * 16) * XS + k0, XS, lane);
#pragma unroll
            for (int n = 0; n < 4; ++n)
                b[n] = load_frag_b(p.W1t, TD, wv * 64 + n * 16, k0, lane);
#pragma unroll
            for (int m = 0; m < 4; ++m)
#pragma unroll
                for (int n = 0; n < 4; ++n)
                    acc[m][n] = WMMA_BF16(a[m], b[n], acc[m][n]);
        }
        // bias
#pragma unroll
        for (int n = 0; n < 4; ++n) {
            int ng = wv * 64 + n * 16 + col;
            float bias = p.enc_b1[ng];
#pragma unroll
            for (int m = 0; m < 4; ++m)
#pragma unroll
                for (int g = 0; g < 8; ++g) acc[m][n][g] += bias;
        }
        // per-row LN statistics: 16-lane shfl reduction + LDS f32 atomics
#pragma unroll
        for (int m = 0; m < 4; ++m)
#pragma unroll
            for (int g = 0; g < 8; ++g) {
                float s = 0.f, q = 0.f;
#pragma unroll
                for (int n = 0; n < 4; ++n) { float v = acc[m][n][g]; s += v; q += v * v; }
#pragma unroll
                for (int off = 1; off < 16; off <<= 1) {
                    s += __shfl_xor(s, off);
                    q += __shfl_xor(q, off);
                }
                if (col == 0) {
                    int row = m * 16 + hsel * 8 + g;
                    atomicAdd(&sSum[row], s);
                    atomicAdd(&sSq[row], q);
                }
            }
        __syncthreads();
        const float inv = 1.f / (float)H1;
#pragma unroll
        for (int m = 0; m < 4; ++m)
#pragma unroll
            for (int g = 0; g < 8; ++g) {
                int row = m * 16 + hsel * 8 + g;
                float mu = sSum[row] * inv;
                float var = sSq[row] * inv - mu * mu;
                float rstd = rsqrtf(var + 1e-5f);
#pragma unroll
                for (int n = 0; n < 4; ++n) {
                    int ng = wv * 64 + n * 16 + col;
                    float v = (acc[m][n][g] - mu) * rstd * p.ln_g[ng] + p.ln_b[ng];
                    sH[row * HS + ng] = (__bf16)gelu_erf(v);
                }
            }
    }
    __syncthreads();

    // ---- Phase 3: aggregated = h @ W2 + b2; M=64 N=256 K=512 ----
    {
        f32x8 acc[4][2];
#pragma unroll
        for (int m = 0; m < 4; ++m)
#pragma unroll
            for (int n = 0; n < 2; ++n) acc[m][n] = zero8();
        for (int kt = 0; kt < H1 / 32; ++kt) {
            const int k0 = kt * 32;
            bf16x16 a[4], b[2];
#pragma unroll
            for (int m = 0; m < 4; ++m)
                a[m] = load_frag_a(sH + (m * 16) * HS + k0, HS, lane);
#pragma unroll
            for (int n = 0; n < 2; ++n)
                b[n] = load_frag_b(p.W2t, H1, wv * 32 + n * 16, k0, lane);
#pragma unroll
            for (int m = 0; m < 4; ++m)
#pragma unroll
                for (int n = 0; n < 2; ++n)
                    acc[m][n] = WMMA_BF16(a[m], b[n], acc[m][n]);
        }
#pragma unroll
        for (int n = 0; n < 2; ++n) {
            int ng = wv * 32 + n * 16 + col;
            float bias = p.enc_b2[ng];
#pragma unroll
            for (int m = 0; m < 4; ++m)
#pragma unroll
                for (int g = 0; g < 8; ++g) {
                    int row = m * 16 + hsel * 8 + g;
                    sA[row * AS + ng] = (__bf16)(acc[m][n][g] + bias);
                }
        }
    }
    __syncthreads();

    // ---- Phase 4: head MLP -> regime probs / initial weights ----
    {
        f32x8 acc[4];
#pragma unroll
        for (int m = 0; m < 4; ++m) acc[m] = zero8();
        for (int kt = 0; kt < D / 32; ++kt) {
            const int k0 = kt * 32;
            bf16x16 b = load_frag_b(p.HW1t, D, wv * 16, k0, lane);
#pragma unroll
            for (int m = 0; m < 4; ++m) {
                bf16x16 a = load_frag_a(sA + (m * 16) * AS + k0, AS, lane);
                acc[m] = WMMA_BF16(a, b, acc[m]);
            }
        }
        int ng = wv * 16 + col;
        float bias = p.head_b1[ng];
#pragma unroll
        for (int m = 0; m < 4; ++m)
#pragma unroll
            for (int g = 0; g < 8; ++g) {
                int row = m * 16 + hsel * 8 + g;
                sT1[row * TS + ng] = (__bf16)gelu_erf(acc[m][g] + bias);
            }
    }
    __syncthreads();
    for (int idx = tid; idx < MT * R; idx += NTHR) {
        int r = idx / R, j = idx - r * R;
        float a = p.head_b2[j];
        for (int k = 0; k < H2; ++k) a += (float)sT1[r * TS + k] * p.head_w2[k * R + j];
        sPr[r * 8 + j] = a;
    }
    __syncthreads();
    if (tid < MT) {
        int r = tid;
        float mx = -1e30f;
#pragma unroll
        for (int j = 0; j < R; ++j) mx = fmaxf(mx, sPr[r * 8 + j]);
        float e[R], sm = 0.f;
#pragma unroll
        for (int j = 0; j < R; ++j) { e[j] = expf(sPr[r * 8 + j] - mx); sm += e[j]; }
        float invs = 1.f / sm;
#pragma unroll
        for (int j = 0; j < R; ++j) sPr[r * 8 + j] = e[j] * invs;
#pragma unroll
        for (int jt = 0; jt < T; ++jt) {
            float w = 0.f;
#pragma unroll
            for (int i = 0; i < R; ++i) w += sPr[r * 8 + i] * p.regime_priors[i * T + jt];
            sW[r * 8 + jt] = w;
        }
    }
    __syncthreads();

    // ---- Phase 5: 3 refinement steps ----
    for (int s = 0; s < S; ++s) {
        {   // build combined[:,0:256] = probs@regime_emb + step_emb[s]  (bf16 in sA)
            int d = tid; // 0..255
            float emb[R];
#pragma unroll
            for (int i = 0; i < R; ++i) emb[i] = p.regime_emb[i * D + d];
            float se = p.step_emb[s * D + d];
            for (int r = 0; r < MT; ++r) {
                float v = se;
#pragma unroll
                for (int i = 0; i < R; ++i) v += sPr[r * 8 + i] * emb[i];
                sA[r * AS + d] = (__bf16)v;
            }
        }
        __syncthreads();
        {   // t1 = gelu(combined @ ref_w1[s] + b1); K split: 256 via WMMA + 5 via VALU
            f32x8 acc[4];
#pragma unroll
            for (int m = 0; m < 4; ++m) acc[m] = zero8();
            const __bf16* wt = p.RW1t + (size_t)s * H2 * D;
            for (int kt = 0; kt < D / 32; ++kt) {
                const int k0 = kt * 32;
                bf16x16 b = load_frag_b(wt, D, wv * 16, k0, lane);
#pragma unroll
                for (int m = 0; m < 4; ++m) {
                    bf16x16 a = load_frag_a(sA + (m * 16) * AS + k0, AS, lane);
                    acc[m] = WMMA_BF16(a, b, acc[m]);
                }
            }
            int ng = wv * 16 + col;
            float bias = p.ref_b1[s * H2 + ng];
            float xw[T];
#pragma unroll
            for (int j = 0; j < T; ++j)
                xw[j] = p.ref_w1[((size_t)s * 261 + D + j) * H2 + ng];
#pragma unroll
            for (int m = 0; m < 4; ++m)
#pragma unroll
                for (int g = 0; g < 8; ++g) {
                    int row = m * 16 + hsel * 8 + g;
                    float v = acc[m][g] + bias;
#pragma unroll
                    for (int j = 0; j < T; ++j) v += sW[row * 8 + j] * xw[j];
                    sT1[row * TS + ng] = (__bf16)gelu_erf(v);
                }
        }
        __syncthreads();
        // residual + gate, parallel over (row, j): 320 tasks on 256 threads
        for (int idx = tid; idx < MT * T; idx += NTHR) {
            int r = idx / T, j = idx - r * T;
            float res = p.ref_b2[s * T + j];
            for (int k = 0; k < H2; ++k)
                res += (float)sT1[r * TS + k] * p.ref_w2[((size_t)s * H2 + k) * T + j];
            float gt = p.gate_b[s * T + j];
            for (int d2 = 0; d2 < D; ++d2)
                gt += (float)sA[r * AS + d2] * p.gate_w[((size_t)s * 261 + d2) * T + j];
#pragma unroll
            for (int i = 0; i < T; ++i)
                gt += sW[r * 8 + i] * p.gate_w[((size_t)s * 261 + D + i) * T + j];
            sTmp[r * 8 + j] = sW[r * 8 + j] + sigmoidf(gt) * res;
        }
        __syncthreads();
        if (tid < MT) {   // tiny per-row softmax
            int r = tid;
            float nv[T], mx = -1e30f;
#pragma unroll
            for (int j = 0; j < T; ++j) { nv[j] = sTmp[r * 8 + j]; mx = fmaxf(mx, nv[j]); }
            float sm = 0.f;
#pragma unroll
            for (int j = 0; j < T; ++j) { nv[j] = expf(nv[j] - mx); sm += nv[j]; }
            float invs = 1.f / sm;
#pragma unroll
            for (int j = 0; j < T; ++j) sW[r * 8 + j] = nv[j] * invs;
        }
        __syncthreads();
    }

    // ---- Phase 6: per-timeframe quality (reuses resident sX) ----
    for (int t = 0; t < T; ++t) {
        {
            f32x8 acc[4];
#pragma unroll
            for (int m = 0; m < 4; ++m) acc[m] = zero8();
            const __bf16* wt = p.QW1t + (size_t)t * H2 * D;
            for (int kt = 0; kt < D / 32; ++kt) {
                const int k0 = kt * 32;
                bf16x16 b = load_frag_b(wt, D, wv * 16, k0, lane);
#pragma unroll
                for (int m = 0; m < 4; ++m) {
                    bf16x16 a = load_frag_a(sX + (m * 16) * XS + t * D + k0, XS, lane);
                    acc[m] = WMMA_BF16(a, b, acc[m]);
                }
            }
            int ng = wv * 16 + col;
            float bias = p.q_b1[t * H2 + ng];
#pragma unroll
            for (int m = 0; m < 4; ++m)
#pragma unroll
                for (int g = 0; g < 8; ++g) {
                    int row = m * 16 + hsel * 8 + g;
                    sT1[row * TS + ng] = (__bf16)gelu_erf(acc[m][g] + bias);
                }
        }
        __syncthreads();
        if (tid < MT) {
            int r = tid;
            float a = p.q_b2[t];
            for (int k = 0; k < H2; ++k) a += (float)sT1[r * TS + k] * p.q_w2[t * H2 + k];
            sQ[r * 8 + t] = sigmoidf(a);
        }
        __syncthreads();
    }

    // ---- Phase 7: final mix + softmax + floor ----
    if (tid < MT) {
        int r = tid;
        float z[T];
#pragma unroll
        for (int j = 0; j < T; ++j) z[j] = p.final_b[j];
#pragma unroll
        for (int i = 0; i < T; ++i) {
            float wvv = sW[r * 8 + i], qv = sQ[r * 8 + i];
#pragma unroll
            for (int j = 0; j < T; ++j)
                z[j] += wvv * p.final_w[i * T + j] + qv * p.final_w[(T + i) * T + j];
        }
        float mx = -1e30f;
#pragma unroll
        for (int j = 0; j < T; ++j) mx = fmaxf(mx, z[j]);
        float sm = 0.f;
#pragma unroll
        for (int j = 0; j < T; ++j) { z[j] = expf(z[j] - mx); sm += z[j]; }
        float invs = 1.f / sm;
#pragma unroll
        for (int j = 0; j < T; ++j)
            p.out[(size_t)(b0 + r) * T + j] = z[j] * invs * (1.f - T * 0.05f) + 0.05f;
    }
}

// Transpose + f32->bf16 conversion: in (K,N) row-major -> out (N,K)
__global__ void k_conv_t(const float* __restrict__ in, __bf16* __restrict__ out, int K, int N)
{
    int idx = blockIdx.x * blockDim.x + threadIdx.x;
    int total = K * N;
    if (idx < total) {
        int n = idx / K, k = idx - n * K;
        out[idx] = (__bf16)in[k * N + n];
    }
}

extern "C" void kernel_launch(void* const* d_in, const int* in_sizes, int n_in,
                              void* d_out, int out_size, void* d_ws, size_t ws_size,
                              hipStream_t stream)
{
    (void)in_sizes; (void)n_in; (void)out_size; (void)ws_size;
    const float* tf       = (const float*)d_in[0];
    const float* enc_w1   = (const float*)d_in[1];
    const float* enc_b1   = (const float*)d_in[2];
    const float* ln_g     = (const float*)d_in[3];
    const float* ln_b     = (const float*)d_in[4];
    const float* enc_w2   = (const float*)d_in[5];
    const float* enc_b2   = (const float*)d_in[6];
    const float* head_w1  = (const float*)d_in[7];
    const float* head_b1  = (const float*)d_in[8];
    const float* head_w2  = (const float*)d_in[9];
    const float* head_b2  = (const float*)d_in[10];
    const float* regime_emb = (const float*)d_in[11];
    const float* q_w1     = (const float*)d_in[12];
    const float* q_b1     = (const float*)d_in[13];
    const float* q_w2     = (const float*)d_in[14];
    const float* q_b2     = (const float*)d_in[15];
    const float* ref_w1   = (const float*)d_in[16];
    const float* ref_b1   = (const float*)d_in[17];
    const float* ref_w2   = (const float*)d_in[18];
    const float* ref_b2   = (const float*)d_in[19];
    const float* gate_w   = (const float*)d_in[20];
    const float* gate_b   = (const float*)d_in[21];
    const float* step_emb = (const float*)d_in[22];
    const float* regime_priors = (const float*)d_in[23];
    const float* final_w  = (const float*)d_in[24];
    const float* final_b  = (const float*)d_in[25];

    // workspace carve: bf16 transposed weights (all L2-resident, ~2.1MB)
    __bf16* W1t  = (__bf16*)d_ws;            // 512 x 1280
    __bf16* W2t  = W1t  + 512 * 1280;        // 256 x 512
    __bf16* HW1t = W2t  + 256 * 512;         // 128 x 256
    __bf16* QW1t = HW1t + 128 * 256;         // 5 x 128 x 256
    __bf16* RW1t = QW1t + 5 * 128 * 256;     // 3 x 128 x 256

    k_conv_t<<<(1280 * 512 + 255) / 256, 256, 0, stream>>>(enc_w1, W1t, 1280, 512);
    k_conv_t<<<(512 * 256 + 255) / 256, 256, 0, stream>>>(enc_w2, W2t, 512, 256);
    k_conv_t<<<(256 * 128 + 255) / 256, 256, 0, stream>>>(head_w1, HW1t, 256, 128);
    for (int t = 0; t < T; ++t)
        k_conv_t<<<(256 * 128 + 255) / 256, 256, 0, stream>>>(
            q_w1 + (size_t)t * 256 * 128, QW1t + (size_t)t * 128 * 256, 256, 128);
    for (int s = 0; s < S; ++s)  // only first 256 K-rows of the (261,128) block
        k_conv_t<<<(256 * 128 + 255) / 256, 256, 0, stream>>>(
            ref_w1 + (size_t)s * 261 * 128, RW1t + (size_t)s * 128 * 256, 256, 128);

    KParams p;
    p.tf = tf; p.enc_b1 = enc_b1; p.ln_g = ln_g; p.ln_b = ln_b; p.enc_b2 = enc_b2;
    p.head_b1 = head_b1; p.head_w2 = head_w2; p.head_b2 = head_b2; p.regime_emb = regime_emb;
    p.q_b1 = q_b1; p.q_w2 = q_w2; p.q_b2 = q_b2;
    p.ref_w1 = ref_w1; p.ref_b1 = ref_b1; p.ref_w2 = ref_w2; p.ref_b2 = ref_b2;
    p.gate_w = gate_w; p.gate_b = gate_b; p.step_emb = step_emb;
    p.regime_priors = regime_priors; p.final_w = final_w; p.final_b = final_b;
    p.W1t = W1t; p.W2t = W2t; p.HW1t = HW1t; p.QW1t = QW1t; p.RW1t = RW1t;
    p.out = (float*)d_out;

    (void)hipFuncSetAttribute((const void*)fused_tfsel,
                              hipFuncAttributeMaxDynamicSharedMemorySize, SMEM_BYTES);
    fused_tfsel<<<Bsz / MT, NTHR, SMEM_BYTES, stream>>>(p);
}
